// SelfAttention_33217277067367
// MI455X (gfx1250) — compile-verified, matrix operand order checked by reading, not run on hardware
//
#include <hip/hip_runtime.h>
#include <hip/hip_bf16.h>

// ---------------------------------------------------------------------------
// CDNA5 (gfx1250) fused self-attention:
//   wave32 + v_wmma_f32_16x16x32_bf16 + async global->LDS double buffering
// ---------------------------------------------------------------------------

typedef __bf16 bf16;
typedef __attribute__((ext_vector_type(16))) __bf16 v16bf;
typedef __attribute__((ext_vector_type(8)))  __bf16 v8bf;
typedef __attribute__((ext_vector_type(8)))  float  v8f;

#define HDIM   4096
#define SEQ    2048
#define NHEADS 32
#define HSZ    128
#define LDT    40        // padded LDS row stride (elements) -> 80B rows, 16B aligned

union FragBf { v16bf v; v8bf h[2]; };

__device__ __forceinline__ v8f wmma_bf16(v16bf a, v16bf b, v8f c) {
    // D = A(16x32 bf16) x B(32x16 bf16) + C(16x16 f32)
    return __builtin_amdgcn_wmma_f32_16x16x32_bf16(
        /*neg_a=*/false, a, /*neg_b=*/false, b,
        /*c_mod=*/(short)0, c, /*reuse_a=*/false, /*reuse_b=*/false);
}

// Generic->LDS byte offset: for __shared__ objects the generic pointer is
// {shared_aperture[63:32], lds_offset[31:0]} (ISA 10.2 aperture rules).
__device__ __forceinline__ unsigned lds_off(const void* p) {
    return (unsigned)(uintptr_t)p;
}

// Async DMA one 16B chunk per lane: global -> LDS (ASYNCcnt-tracked).
__device__ __forceinline__ void async_ld_b128(unsigned ldsaddr, const bf16* gptr) {
    asm volatile("global_load_async_to_lds_b128 %0, %1, off"
                 :: "v"(ldsaddr), "v"(gptr) : "memory");
}
__device__ __forceinline__ void wait_async0() {
    asm volatile("s_wait_asynccnt 0x0" ::: "memory");
}

// Load a 16x32 (rows x K) bf16 fragment from a row-major global matrix.
// Per-lane layout (ISA 7.12.2, 16-bit A / B): lanes 0-15 hold K 0..7 & 16..23,
// lanes 16-31 hold K 8..15 & 24..31, for row/col = lane&15.
__device__ __forceinline__ v16bf load_gfrag(const bf16* __restrict__ base,
                                            int ld, int r0, int k0, int lane) {
    int r  = r0 + (lane & 15);
    int kk = k0 + ((lane >> 4) << 3);
    const bf16* p = base + (size_t)r * ld + kk;
    FragBf f;
    f.h[0] = *(const v8bf*)(p);
    f.h[1] = *(const v8bf*)(p + 16);
    return f.v;
}

// ---------------------------------------------------------------------------
// GEMM: Y[m,n] = sum_k X[m,k] * W[n,k]   (i.e. X @ W^T), M=N=K=4096, bf16.
// MODE 0: Q-proj + RoPE     -> Y row-major [4096,4096]
// MODE 1: K-proj + RoPE     -> Y row-major [4096,4096]
// MODE 2: V-proj, store V^T -> Y [(b*32+h)*128 + d][s]  (row stride 2048)
// MODE 3: O-proj            -> Y row-major [4096,4096]
// Block: 256 threads (8 waves). Tile 256M x 128N, K-step 32.
// Double-buffered LDS fed by async global->LDS DMA; one barrier per K-step.
// Each wave: 32 M-rows (2 A-frags) x 128 N -> 16 WMMA per K-step.
// ---------------------------------------------------------------------------
template <int MODE>
__global__ __launch_bounds__(256)
void gemm_rope_kernel(const bf16* __restrict__ X,
                      const bf16* __restrict__ W,
                      bf16* __restrict__ Y,
                      const bf16* __restrict__ sinT,
                      const bf16* __restrict__ cosT,
                      const int* __restrict__ startp) {
    __shared__ bf16 Xs[2][256 * LDT];   // 2 x 20KB
    __shared__ bf16 Ws[2][128 * LDT];   // 2 x 10KB

    const int tid  = threadIdx.x;
    const int lane = tid & 31;
    const int wave = tid >> 5;
    const int nBase = blockIdx.x * 128;
    const int mBase = blockIdx.y * 256;

    v8f acc0[8] = {};
    v8f acc1[8] = {};

    const int akk = (lane >> 4) << 3;

    // issue async copies of K-step `ks` into buffer `buf`
    auto stage = [&](int ks, int buf) {
        const int k0 = ks * 32;
#pragma unroll
        for (int i = 0; i < 4; ++i) {                 // X: 256x32 = 1024 chunks
            int c = tid + i * 256;
            int row = c >> 2, c8 = (c & 3) << 3;
            async_ld_b128(lds_off(&Xs[buf][row * LDT + c8]),
                          X + (size_t)(mBase + row) * HDIM + k0 + c8);
        }
#pragma unroll
        for (int i = 0; i < 2; ++i) {                 // W: 128x32 = 512 chunks
            int c = tid + i * 256;
            int row = c >> 2, c8 = (c & 3) << 3;
            async_ld_b128(lds_off(&Ws[buf][row * LDT + c8]),
                          W + (size_t)(nBase + row) * HDIM + k0 + c8);
        }
    };

    stage(0, 0);
    wait_async0();
    __syncthreads();

    const int NK = HDIM / 32;
    for (int ks = 0; ks < NK; ++ks) {
        const int cur = ks & 1;
        if (ks + 1 < NK) stage(ks + 1, cur ^ 1);      // DMA next tile in flight

        const int ar0 = wave * 32 + (lane & 15);
        FragBf a0, a1;
        a0.h[0] = *(const v8bf*)&Xs[cur][ar0 * LDT + akk];
        a0.h[1] = *(const v8bf*)&Xs[cur][ar0 * LDT + akk + 16];
        a1.h[0] = *(const v8bf*)&Xs[cur][(ar0 + 16) * LDT + akk];
        a1.h[1] = *(const v8bf*)&Xs[cur][(ar0 + 16) * LDT + akk + 16];
#pragma unroll
        for (int nt = 0; nt < 8; ++nt) {
            FragBf b;
            int br = nt * 16 + (lane & 15);
            b.h[0] = *(const v8bf*)&Ws[cur][br * LDT + akk];
            b.h[1] = *(const v8bf*)&Ws[cur][br * LDT + akk + 16];
            acc0[nt] = wmma_bf16(a0.v, b.v, acc0[nt]);
            acc1[nt] = wmma_bf16(a1.v, b.v, acc1[nt]);
        }
        wait_async0();                                // next tile fully in LDS
        __syncthreads();
    }

    // ---- epilogue: C-layout element (reg r, lane) -> row m, col n ----
    const int start   = startp[0];
    const int halfsel = lane >> 4;
#pragma unroll
    for (int mi = 0; mi < 2; ++mi) {
        v8f* accp = mi ? acc1 : acc0;
#pragma unroll
        for (int nt = 0; nt < 8; ++nt) {
#pragma unroll
            for (int r = 0; r < 8; ++r) {
                int   m = mBase + wave * 32 + mi * 16 + r + halfsel * 8;
                int   n = nBase + nt * 16 + (lane & 15);
                float v = accp[nt][r];
                if (MODE == 0 || MODE == 1) {               // RoPE
                    int   pos = (m & (SEQ - 1)) + start;
                    int   d   = n & (HSZ - 1);
                    int   p   = d >> 1;
                    float sn  = (float)sinT[pos * (HSZ / 2) + p];
                    float cs  = (float)cosT[pos * (HSZ / 2) + p];
                    float other = __shfl_xor(v, 1);         // even<->odd partner
                    float o = ((d & 1) == 0) ? (cs * v - sn * other)
                                             : (sn * other + cs * v);
                    Y[(size_t)m * HDIM + n] = (bf16)o;
                } else if (MODE == 2) {                     // V stored transposed
                    int bb = m >> 11;                       // batch
                    int s  = m & (SEQ - 1);
                    int hh = n >> 7;                        // head
                    int d  = n & (HSZ - 1);
                    Y[(((size_t)bb * NHEADS + hh) * HSZ + d) * SEQ + s] = (bf16)v;
                } else {                                    // plain store (O-proj)
                    Y[(size_t)m * HDIM + n] = (bf16)v;
                }
            }
        }
    }
}

// ---------------------------------------------------------------------------
// Flash attention: grid = (16 q-tiles, 64 b*h), 256 threads (8 waves).
// Each wave owns 16 q-rows and the full d=128 accumulator (8 f32 C tiles).
// Scores via WMMA (Q-frag x K-frag), online softmax with xor-shuffle row
// reductions, P re-laid out C->A through a per-wave LDS strip, P@V via WMMA
// against the pre-transposed V^T.
// ---------------------------------------------------------------------------
__global__ __launch_bounds__(256)
void attn_kernel(const bf16* __restrict__ Q,
                 const bf16* __restrict__ K,
                 const bf16* __restrict__ Vt,
                 bf16* __restrict__ Ctx) {
    __shared__ bf16 Ps[8 * 16 * LDT];          // per-wave 16x32 (padded)

    const int tid  = threadIdx.x;
    const int lane = tid & 31;
    const int wave = tid >> 5;
    const int qt   = blockIdx.x;               // 0..15
    const int bh   = blockIdx.y;               // 0..63
    const int b    = bh >> 5;
    const int h    = bh & (NHEADS - 1);

    const bf16* qbase = Q  + (size_t)b * SEQ * HDIM + h * HSZ;
    const bf16* kbase = K  + (size_t)b * SEQ * HDIM + h * HSZ;
    const bf16* vbase = Vt + (size_t)bh * HSZ * SEQ;

    const int m0 = qt * 128 + wave * 16;       // this wave's q-row base

    FragBf qf[4];
#pragma unroll
    for (int dk = 0; dk < 4; ++dk)
        qf[dk].v = load_gfrag(qbase, HDIM, m0, dk * 32, lane);

    v8f   acc[8] = {};
    float mstat[8], lstat[8];
#pragma unroll
    for (int r = 0; r < 8; ++r) { mstat[r] = -1e30f; lstat[r] = 0.0f; }

    bf16* pw = &Ps[wave * 16 * LDT];
    const float scale = 0.0883883476483184f;   // 1/sqrt(128)
    const int   colL  = lane & 15;
    const int   akk   = (lane >> 4) << 3;
    const int   nkt   = qt * 4 + 4;            // causal: key tiles of 32

    for (int kt = 0; kt < nkt; ++kt) {
        const int kb = kt * 32;

        if (kt + 1 < nkt) {                    // prefetch next K/V tile rows
            __builtin_prefetch(kbase + (size_t)(kb + 32 + (lane & 15)) * HDIM, 0, 1);
            __builtin_prefetch(vbase + (size_t)((lane & 15) * 8 + (lane >> 4) * 4) * SEQ
                                     + kb + 32, 0, 1);
        }

        // S = Q @ K^T for 32 keys (two 16-key N tiles)
        v8f s0 = {}, s1 = {};
#pragma unroll
        for (int dk = 0; dk < 4; ++dk) {
            v16bf b0 = load_gfrag(kbase, HDIM, kb,      dk * 32, lane);
            s0 = wmma_bf16(qf[dk].v, b0, s0);
            v16bf b1 = load_gfrag(kbase, HDIM, kb + 16, dk * 32, lane);
            s1 = wmma_bf16(qf[dk].v, b1, s1);
        }

        // online softmax update + emit P (bf16) into this wave's LDS strip
#pragma unroll
        for (int r = 0; r < 8; ++r) {
            int   qrow = m0 + r + (lane >> 4) * 8;
            float v0 = s0[r] * scale;
            float v1 = s1[r] * scale;
            if (kb + colL > qrow)      v0 = -1e30f;   // causal mask
            if (kb + 16 + colL > qrow) v1 = -1e30f;

            float mx = fmaxf(v0, v1);
            mx = fmaxf(mx, __shfl_xor(mx, 1));
            mx = fmaxf(mx, __shfl_xor(mx, 2));
            mx = fmaxf(mx, __shfl_xor(mx, 4));
            mx = fmaxf(mx, __shfl_xor(mx, 8));
            float mnew = fmaxf(mstat[r], mx);
            float corr = __expf(mstat[r] - mnew);
            float p0   = __expf(v0 - mnew);
            float p1   = __expf(v1 - mnew);

            float rs = p0 + p1;
            rs += __shfl_xor(rs, 1);
            rs += __shfl_xor(rs, 2);
            rs += __shfl_xor(rs, 4);
            rs += __shfl_xor(rs, 8);
            lstat[r] = lstat[r] * corr + rs;
            mstat[r] = mnew;
#pragma unroll
            for (int nt = 0; nt < 8; ++nt) acc[nt][r] *= corr;

            int prow = r + (lane >> 4) * 8;
            pw[prow * LDT + colL]      = (bf16)p0;
            pw[prow * LDT + 16 + colL] = (bf16)p1;
        }
        asm volatile("s_wait_dscnt 0x0" ::: "memory");

        // reload P in A-fragment layout (per-wave LDS, no barrier needed)
        FragBf pa;
        pa.h[0] = *(const v8bf*)&pw[(lane & 15) * LDT + akk];
        pa.h[1] = *(const v8bf*)&pw[(lane & 15) * LDT + akk + 16];

        // acc += P(16x32) @ V(32x128) using pre-transposed V^T
#pragma unroll
        for (int nt = 0; nt < 8; ++nt) {
            v16bf bv = load_gfrag(vbase, SEQ, nt * 16, kb, lane);
            acc[nt] = wmma_bf16(pa.v, bv, acc[nt]);
        }
    }

    // normalize + store ctx in (b, s, h*128+d) layout
#pragma unroll
    for (int nt = 0; nt < 8; ++nt) {
#pragma unroll
        for (int r = 0; r < 8; ++r) {
            int   s = m0 + r + (lane >> 4) * 8;
            int   d = nt * 16 + (lane & 15);
            float o = acc[nt][r] / lstat[r];
            Ctx[((size_t)b * SEQ + s) * HDIM + h * HSZ + d] = (bf16)o;
        }
    }
}

// ---------------------------------------------------------------------------
extern "C" void kernel_launch(void* const* d_in, const int* in_sizes, int n_in,
                              void* d_out, int out_size, void* d_ws, size_t ws_size,
                              hipStream_t stream) {
    const bf16* x    = (const bf16*)d_in[0];
    const bf16* wq   = (const bf16*)d_in[1];
    const bf16* wk   = (const bf16*)d_in[2];
    const bf16* wv   = (const bf16*)d_in[3];
    const bf16* wo   = (const bf16*)d_in[4];
    const bf16* sinT = (const bf16*)d_in[5];
    const bf16* cosT = (const bf16*)d_in[6];
    // d_in[7] = mask (causal, computed analytically on device)
    const int*  start = (const int*)d_in[8];
    bf16* out = (bf16*)d_out;

    char*  ws   = (char*)d_ws;
    size_t MATB = (size_t)HDIM * HDIM * sizeof(bf16);   // 32 MB
    bf16* q   = (bf16*)(ws);
    bf16* k   = (bf16*)(ws + MATB);
    bf16* vt  = (bf16*)(ws + 2 * MATB);
    bf16* ctx = (bf16*)(ws + 3 * MATB);

    dim3 gg(HDIM / 128, HDIM / 256);   // 32 x 16
    dim3 bb(256);
    gemm_rope_kernel<0><<<gg, bb, 0, stream>>>(x, wq, q,  sinT, cosT, start);
    gemm_rope_kernel<1><<<gg, bb, 0, stream>>>(x, wk, k,  sinT, cosT, start);
    gemm_rope_kernel<2><<<gg, bb, 0, stream>>>(x, wv, vt, sinT, cosT, start);
    attn_kernel<<<dim3(SEQ / 128, 2 * NHEADS), bb, 0, stream>>>(q, k, vt, ctx);
    gemm_rope_kernel<3><<<gg, bb, 0, stream>>>(ctx, wo, out, sinT, cosT, start);
}
